// GraphPolicyNetwork_53772990546392
// MI455X (gfx1250) — matrix operand order repaired and checked by reading, not compiled, not run on hardware
//
#include <hip/hip_runtime.h>
#include <hip/hip_bf16.h>

// ---------------- problem constants ----------------
constexpr int N        = 4096;
constexpr int NODE_DIM = 384;
constexpr int HID      = 256;
constexpr int HEADS    = 4;
constexpr int D        = HID * HEADS;   // 1024
constexpr int STATE    = 3 * D;         // 3072

typedef __bf16 bf16;
typedef __attribute__((ext_vector_type(16))) __bf16 v16bf;
typedef __attribute__((ext_vector_type(8)))  __bf16 v8bf;
typedef __attribute__((ext_vector_type(8)))  float  v8f;

union bfrag { v16bf v; v8bf h[2]; };

// float -> bf16 bits, round-to-nearest-even
__device__ __forceinline__ unsigned short f2bfu(float f) {
    union { float f; unsigned u; } x; x.f = f;
    return (unsigned short)((x.u + 0x7FFFu + ((x.u >> 16) & 1u)) >> 16);
}
__device__ __forceinline__ bf16 f2bf(float f) {
    unsigned short s = f2bfu(f);
    bf16 out;
    __builtin_memcpy(&out, &s, 2);
    return out;
}
__device__ __forceinline__ unsigned pack2(float lo, float hi) {
    return (unsigned)f2bfu(lo) | ((unsigned)f2bfu(hi) << 16);
}

#define WMMA_BF16(A, B, C) \
    __builtin_amdgcn_wmma_f32_16x16x32_bf16(false, (A), false, (B), (short)0, (C), false, false)

// A fragment: row-major LDS tile; two contiguous 16B runs -> 2x ds_load_b128
__device__ __forceinline__ v16bf load_afrag(const bf16* rowp, int hf) {
    bfrag a;
    a.h[0] = *(const v8bf*)(rowp + hf * 8);
    a.h[1] = *(const v8bf*)(rowp + 16 + hf * 8);
    return a.v;
}
// B fragment: fragment-major LDS tile BsT[col][k]; two contiguous 16B runs -> 2x ds_load_b128
__device__ __forceinline__ v16bf load_bfrag(const bf16* colp, int hf) {
    bfrag b;
    b.h[0] = *(const v8bf*)(colp + hf * 8);
    b.h[1] = *(const v8bf*)(colp + 16 + hf * 8);
    return b.v;
}

// =====================================================================
// 1) Wh[h] = X @ W[h]   (X: N x K fp32, W: HEADS x K x HID)
//    block tile 64x128, K-step 32, 8 waves (2x4), 4 WMMA / wave / step
// =====================================================================
__global__ __launch_bounds__(256) void k_gemm_wh(
    const float* __restrict__ X, int K, const float* __restrict__ W,
    float* __restrict__ C)
{
    __shared__ bf16 As[64][32];     // row-major (A operand)
    __shared__ bf16 BsT[128][32];   // fragment-major: BsT[col][k]
    const int tid  = threadIdx.x;
    const int lane = tid & 31, wave = tid >> 5;
    const int m0   = blockIdx.x * 64;
    const int n0   = blockIdx.y * 128;
    const int head = blockIdx.z;
    const float* Wp = W + (size_t)head * K * HID;
    float*       Cp = C + (size_t)head * N * HID;

    const int wm = wave & 1, wn = wave >> 1;
    const int r16 = lane & 15, hf = lane >> 4;

    v8f acc00 = {}, acc01 = {}, acc10 = {}, acc11 = {};
    for (int k0 = 0; k0 < K; k0 += 32) {
        // A tile: 512 float4 loads (b128), packed 8B LDS stores
        #pragma unroll
        for (int e = tid; e < 512; e += 256) {
            int r = e >> 3, c4 = (e & 7) * 4;
            const float4 x = *(const float4*)&X[(size_t)(m0 + r) * K + k0 + c4];
            uint2 pk; pk.x = pack2(x.x, x.y); pk.y = pack2(x.z, x.w);
            *(uint2*)&As[r][c4] = pk;
        }
        // B tile: 1024 float4 loads (b128), transposed b16 LDS stores
        #pragma unroll
        for (int e = tid; e < 1024; e += 256) {
            int c4 = (e & 31) * 4, r = e >> 5;
            const float4 x = *(const float4*)&Wp[(size_t)(k0 + r) * HID + n0 + c4];
            BsT[c4 + 0][r] = f2bf(x.x);
            BsT[c4 + 1][r] = f2bf(x.y);
            BsT[c4 + 2][r] = f2bf(x.z);
            BsT[c4 + 3][r] = f2bf(x.w);
        }
        if (k0 + 32 < K) {                                  // global_prefetch_b8
            __builtin_prefetch(&X[(size_t)(m0 + (tid & 63)) * K + k0 + 32], 0, 1);
            __builtin_prefetch(&Wp[(size_t)(k0 + 32 + (tid >> 3)) * HID + n0 + (tid & 7) * 16], 0, 1);
        }
        __syncthreads();

        v16bf a0 = load_afrag(&As[wm * 32 + r16][0], hf);
        v16bf a1 = load_afrag(&As[wm * 32 + 16 + r16][0], hf);
        v16bf b0 = load_bfrag(&BsT[wn * 32 + r16][0], hf);
        v16bf b1 = load_bfrag(&BsT[wn * 32 + 16 + r16][0], hf);
        acc00 = WMMA_BF16(a0, b0, acc00);
        acc01 = WMMA_BF16(a0, b1, acc01);
        acc10 = WMMA_BF16(a1, b0, acc10);
        acc11 = WMMA_BF16(a1, b1, acc11);
        __syncthreads();
    }
    #pragma unroll
    for (int v = 0; v < 8; ++v) {   // D layout: row = v + 8*hf, col = r16
        int r0 = m0 + wm * 32 + v + 8 * hf;
        int c0 = n0 + wn * 32 + r16;
        Cp[(size_t)r0 * HID + c0]             = acc00[v];
        Cp[(size_t)r0 * HID + c0 + 16]        = acc01[v];
        Cp[(size_t)(r0 + 16) * HID + c0]      = acc10[v];
        Cp[(size_t)(r0 + 16) * HID + c0 + 16] = acc11[v];
    }
}

// =====================================================================
// 2) src/dst attention scalars — one wave per node, lane-parallel float4
// =====================================================================
__global__ __launch_bounds__(256) void k_src_dst(
    const float* __restrict__ Wh, const float* __restrict__ a,
    float* __restrict__ srcv, float* __restrict__ dstv)
{
    const int lane = threadIdx.x & 31, wave = threadIdx.x >> 5;
    const int n = blockIdx.x * 8 + wave;
    const int h = blockIdx.y;
    const float* w  = Wh + ((size_t)h * N + n) * HID;
    const float* ai = a + (size_t)h * 2 * HID;
    const float* aj = ai + HID;
    float si = 0.f, sj = 0.f;
    #pragma unroll
    for (int i = 0; i < 2; ++i) {
        int o = (lane + 32 * i) * 4;
        float4 wv = *(const float4*)&w[o];
        float4 av = *(const float4*)&ai[o];
        float4 bv = *(const float4*)&aj[o];
        si += wv.x * av.x + wv.y * av.y + wv.z * av.z + wv.w * av.w;
        sj += wv.x * bv.x + wv.y * bv.y + wv.z * bv.z + wv.w * bv.w;
    }
    #pragma unroll
    for (int off = 16; off > 0; off >>= 1) {
        si += __shfl_xor(si, off, 32);
        sj += __shfl_xor(sj, off, 32);
    }
    if (lane == 0) {
        srcv[(size_t)h * N + n] = si;
        dstv[(size_t)h * N + n] = sj;
    }
}

// =====================================================================
// 3) attention: out = softmax(mask(leaky(src+dst))) @ Wh
//    32 rows / block, m-step 32, Wh staged to LDS, 4 WMMA / wave / step
// =====================================================================
__global__ __launch_bounds__(256) void k_attn(
    const float* __restrict__ Wh, const float* __restrict__ srcv,
    const float* __restrict__ dstv, const float* __restrict__ adj,
    float* __restrict__ G)
{
    __shared__ float rowmax[32];
    __shared__ float rowsum[32];
    __shared__ bf16  Ps[32][32];      // P tile (A operand)
    __shared__ bf16  WhS[256][32];    // Wh K-tile, fragment-major (B operand)
    const int tid = threadIdx.x, lane = tid & 31, wave = tid >> 5;
    const int head = blockIdx.y;
    const int n0   = blockIdx.x * 32;
    const float* srcH = srcv + (size_t)head * N;
    const float* dstH = dstv + (size_t)head * N;
    const float* WhH  = Wh + (size_t)head * N * HID;

    // ---- pass 1: masked row maxima (wave w owns rows w+8k), float4 scans ----
    #pragma unroll
    for (int rr = 0; rr < 4; ++rr) {
        int r = wave + 8 * rr;
        int n = n0 + r;
        float s = srcH[n];
        float mx = -3.0e38f;
        for (int m = lane * 4; m < N; m += 128) {
            const float4 a4 = *(const float4*)&adj[(size_t)n * N + m];
            const float4 d4 = *(const float4*)&dstH[m];
            if (a4.x > 0.f) { float e = s + d4.x; mx = fmaxf(mx, (e > 0.f) ? e : 0.2f * e); }
            if (a4.y > 0.f) { float e = s + d4.y; mx = fmaxf(mx, (e > 0.f) ? e : 0.2f * e); }
            if (a4.z > 0.f) { float e = s + d4.z; mx = fmaxf(mx, (e > 0.f) ? e : 0.2f * e); }
            if (a4.w > 0.f) { float e = s + d4.w; mx = fmaxf(mx, (e > 0.f) ? e : 0.2f * e); }
        }
        #pragma unroll
        for (int off = 16; off > 0; off >>= 1)
            mx = fmaxf(mx, __shfl_xor(mx, off, 32));
        if (lane == 0) rowmax[r] = mx;
    }
    if (tid < 32) rowsum[tid] = 0.f;
    __syncthreads();

    // ---- pass 2 ----
    const int r16 = lane & 15, hf = lane >> 4;
    const int col0 = wave * 32;
    const int pr = tid >> 3, pc4 = (tid & 7) * 4;   // this thread's P-tile slot
    v8f acc00 = {}, acc01 = {}, acc10 = {}, acc11 = {};
    for (int m0 = 0; m0 < N; m0 += 32) {
        {   // P tile: one float4 of adj per thread, one rowsum atomic per thread
            int n = n0 + pr;
            const float4 a4 = *(const float4*)&adj[(size_t)n * N + m0 + pc4];
            const float4 d4 = *(const float4*)&dstH[m0 + pc4];
            float s = srcH[n], mxr = rowmax[pr];
            float p0 = 0.f, p1 = 0.f, p2 = 0.f, p3 = 0.f;
            if (a4.x > 0.f) { float e = s + d4.x; e = (e > 0.f) ? e : 0.2f * e; p0 = __expf(e - mxr); }
            if (a4.y > 0.f) { float e = s + d4.y; e = (e > 0.f) ? e : 0.2f * e; p1 = __expf(e - mxr); }
            if (a4.z > 0.f) { float e = s + d4.z; e = (e > 0.f) ? e : 0.2f * e; p2 = __expf(e - mxr); }
            if (a4.w > 0.f) { float e = s + d4.w; e = (e > 0.f) ? e : 0.2f * e; p3 = __expf(e - mxr); }
            float psum = p0 + p1 + p2 + p3;
            if (psum > 0.f) atomicAdd(&rowsum[pr], psum);   // ds_add_f32, 1/thread
            uint2 pk; pk.x = pack2(p0, p1); pk.y = pack2(p2, p3);
            *(uint2*)&Ps[pr][pc4] = pk;
        }
        // Wh tile: 2048 float4 loads (b128), transposed b16 LDS stores
        #pragma unroll
        for (int e = tid; e < 2048; e += 256) {
            int c4 = (e & 63) * 4, r = e >> 6;
            const float4 x = *(const float4*)&WhH[(size_t)(m0 + r) * HID + c4];
            WhS[c4 + 0][r] = f2bf(x.x);
            WhS[c4 + 1][r] = f2bf(x.y);
            WhS[c4 + 2][r] = f2bf(x.z);
            WhS[c4 + 3][r] = f2bf(x.w);
        }
        __syncthreads();

        v16bf a0 = load_afrag(&Ps[r16][0], hf);
        v16bf a1 = load_afrag(&Ps[16 + r16][0], hf);
        v16bf b0 = load_bfrag(&WhS[col0 + r16][0], hf);
        v16bf b1 = load_bfrag(&WhS[col0 + 16 + r16][0], hf);
        acc00 = WMMA_BF16(a0, b0, acc00);
        acc01 = WMMA_BF16(a0, b1, acc01);
        acc10 = WMMA_BF16(a1, b0, acc10);
        acc11 = WMMA_BF16(a1, b1, acc11);
        __syncthreads();
    }
    #pragma unroll
    for (int v = 0; v < 8; ++v) {
        int r = v + 8 * hf;
        float s0 = rowsum[r], s1 = rowsum[16 + r];
        float inv0 = (s0 > 0.f) ? 1.f / s0 : 0.f;      // nan_to_num semantics
        float inv1 = (s1 > 0.f) ? 1.f / s1 : 0.f;
        float* o0 = &G[(size_t)(n0 + r) * D + (size_t)head * HID + col0 + r16];
        float* o1 = &G[(size_t)(n0 + 16 + r) * D + (size_t)head * HID + col0 + r16];
        o0[0]  = acc00[v] * inv0;
        o0[16] = acc01[v] * inv0;
        o1[0]  = acc10[v] * inv1;
        o1[16] = acc11[v] * inv1;
    }
}

// =====================================================================
// 4) fused LayerNorm + ELU, one row (D=1024) per block
// =====================================================================
__global__ __launch_bounds__(256) void k_ln_elu(
    const float* __restrict__ Gin, const float* __restrict__ gamma,
    const float* __restrict__ beta, float* __restrict__ H)
{
    __shared__ float warpsum[8];
    __shared__ float stat[2];
    const int row = blockIdx.x, tid = threadIdx.x;
    const int lane = tid & 31, wave = tid >> 5;
    float x[4];
    float s = 0.f;
    #pragma unroll
    for (int i = 0; i < 4; ++i) {
        x[i] = Gin[(size_t)row * D + tid + 256 * i];
        s += x[i];
    }
    #pragma unroll
    for (int off = 16; off > 0; off >>= 1) s += __shfl_xor(s, off, 32);
    if (lane == 0) warpsum[wave] = s;
    __syncthreads();
    if (tid == 0) {
        float t = 0.f;
        for (int i = 0; i < 8; ++i) t += warpsum[i];
        stat[0] = t / (float)D;
    }
    __syncthreads();
    float mu = stat[0];
    float v = 0.f;
    #pragma unroll
    for (int i = 0; i < 4; ++i) { float d = x[i] - mu; v += d * d; }
    #pragma unroll
    for (int off = 16; off > 0; off >>= 1) v += __shfl_xor(v, off, 32);
    __syncthreads();
    if (lane == 0) warpsum[wave] = v;
    __syncthreads();
    if (tid == 0) {
        float t = 0.f;
        for (int i = 0; i < 8; ++i) t += warpsum[i];
        stat[1] = rsqrtf(t / (float)D + 1e-5f);
    }
    __syncthreads();
    float rstd = stat[1];
    #pragma unroll
    for (int i = 0; i < 4; ++i) {
        int idx = tid + 256 * i;
        float y = (x[i] - mu) * rstd * gamma[idx] + beta[idx];
        H[(size_t)row * D + idx] = (y > 0.f) ? y : (__expf(y) - 1.f);  // ELU
    }
}

// =====================================================================
// 5) claim projection + visited aggregation
// =====================================================================
__global__ __launch_bounds__(256) void k_claim_visited(
    const float* __restrict__ claim, const float* __restrict__ Wc,
    const float* __restrict__ bc, const float* __restrict__ H2,
    const float* __restrict__ mask, float* __restrict__ cp,
    float* __restrict__ vag)
{
    int d = blockIdx.x * 256 + threadIdx.x;
    float c = bc[d];
    for (int k = 0; k < NODE_DIM; ++k) c += claim[k] * Wc[(size_t)k * D + d];
    cp[d] = c;
    float s = 0.f, cnt = 0.f;
    for (int n = 0; n < N; ++n) {
        float mk = mask[n];
        cnt += mk;
        s += H2[(size_t)n * D + d] * mk;
    }
    vag[d] = s / fmaxf(cnt, 1.f);
}

// =====================================================================
// 6) edge MLP layer 1: hidden = relu(NS @ Wa1 + ba1), NS gathered on the fly
//    K-chunks never straddle segment boundaries (32 | 1024), so the segment
//    is uniform per K-step -> float4 staging throughout
// =====================================================================
__global__ __launch_bounds__(256) void k_edge_gemm(
    const float* __restrict__ cp, const float* __restrict__ H2,
    const float* __restrict__ vag, const float* __restrict__ Wa1,
    const float* __restrict__ ba1, float* __restrict__ hid)
{
    __shared__ bf16 As[64][32];
    __shared__ bf16 BsT[128][32];
    const int tid  = threadIdx.x;
    const int lane = tid & 31, wave = tid >> 5;
    const int m0   = blockIdx.x * 64;
    const int n0   = blockIdx.y * 128;
    const int wm = wave & 1, wn = wave >> 1;
    const int r16 = lane & 15, hf = lane >> 4;

    v8f acc00 = {}, acc01 = {}, acc10 = {}, acc11 = {};
    for (int k0 = 0; k0 < STATE; k0 += 32) {
        const int seg = (k0 < D) ? 0 : ((k0 < 2 * D) ? 1 : 2);
        #pragma unroll
        for (int e = tid; e < 512; e += 256) {
            int r = e >> 3, c4 = (e & 7) * 4;
            float4 x;
            if (seg == 0)      x = *(const float4*)&cp[k0 + c4];
            else if (seg == 2) x = *(const float4*)&vag[k0 - 2 * D + c4];
            else               x = *(const float4*)&H2[(size_t)(m0 + r) * D + (k0 - D) + c4];
            uint2 pk; pk.x = pack2(x.x, x.y); pk.y = pack2(x.z, x.w);
            *(uint2*)&As[r][c4] = pk;
        }
        #pragma unroll
        for (int e = tid; e < 1024; e += 256) {
            int c4 = (e & 31) * 4, r = e >> 5;
            const float4 x = *(const float4*)&Wa1[(size_t)(k0 + r) * HID + n0 + c4];
            BsT[c4 + 0][r] = f2bf(x.x);
            BsT[c4 + 1][r] = f2bf(x.y);
            BsT[c4 + 2][r] = f2bf(x.z);
            BsT[c4 + 3][r] = f2bf(x.w);
        }
        if (k0 + 32 < STATE)
            __builtin_prefetch(&Wa1[(size_t)(k0 + 32 + (tid >> 3)) * HID + n0 + (tid & 7) * 16], 0, 1);
        __syncthreads();

        v16bf a0 = load_afrag(&As[wm * 32 + r16][0], hf);
        v16bf a1 = load_afrag(&As[wm * 32 + 16 + r16][0], hf);
        v16bf b0 = load_bfrag(&BsT[wn * 32 + r16][0], hf);
        v16bf b1 = load_bfrag(&BsT[wn * 32 + 16 + r16][0], hf);
        acc00 = WMMA_BF16(a0, b0, acc00);
        acc01 = WMMA_BF16(a0, b1, acc01);
        acc10 = WMMA_BF16(a1, b0, acc10);
        acc11 = WMMA_BF16(a1, b1, acc11);
        __syncthreads();
    }
    #pragma unroll
    for (int v = 0; v < 8; ++v) {
        int r0 = m0 + wm * 32 + v + 8 * hf;
        int c0 = n0 + wn * 32 + r16;
        float y00 = acc00[v] + ba1[c0];
        float y01 = acc01[v] + ba1[c0 + 16];
        float y10 = acc10[v] + ba1[c0];
        float y11 = acc11[v] + ba1[c0 + 16];
        hid[(size_t)r0 * HID + c0]             = (y00 > 0.f) ? y00 : 0.f;
        hid[(size_t)r0 * HID + c0 + 16]        = (y01 > 0.f) ? y01 : 0.f;
        hid[(size_t)(r0 + 16) * HID + c0]      = (y10 > 0.f) ? y10 : 0.f;
        hid[(size_t)(r0 + 16) * HID + c0 + 16] = (y11 > 0.f) ? y11 : 0.f;
    }
}

// =====================================================================
// 7) edge MLP layer 2 + neighbor masking -> out[0..N-1]
// =====================================================================
__global__ __launch_bounds__(256) void k_edge_final(
    const float* __restrict__ hid, const float* __restrict__ Wa2,
    const float* __restrict__ ba2, const float* __restrict__ adj,
    const int* __restrict__ curp, float* __restrict__ out)
{
    int n = blockIdx.x * 256 + threadIdx.x;
    float s = ba2[0];
    const float* hrow = hid + (size_t)n * HID;
    for (int k = 0; k < HID; ++k) s += hrow[k] * Wa2[k];
    int cur = curp[0];
    out[n] = (adj[(size_t)cur * N + n] > 0.f) ? s : -1.0e9f;
}

// =====================================================================
// 8) stop-score MLP -> out[N]
// =====================================================================
__global__ __launch_bounds__(256) void k_stop(
    const float* __restrict__ cp, const float* __restrict__ H2,
    const float* __restrict__ vag, const int* __restrict__ curp,
    const float* __restrict__ Ws1, const float* __restrict__ bs1,
    const float* __restrict__ Ws2, const float* __restrict__ bs2,
    float* __restrict__ out)
{
    __shared__ float red[8];
    int tid = threadIdx.x, lane = tid & 31, wave = tid >> 5;
    int cur = curp[0];
    const float* hc = H2 + (size_t)cur * D;
    float hsum = bs1[tid];
    for (int k = 0; k < D; ++k) hsum += cp[k]  * Ws1[(size_t)k * HID + tid];
    for (int k = 0; k < D; ++k) hsum += hc[k]  * Ws1[(size_t)(D + k) * HID + tid];
    for (int k = 0; k < D; ++k) hsum += vag[k] * Ws1[(size_t)(2 * D + k) * HID + tid];
    float hv = (hsum > 0.f) ? hsum : 0.f;
    float p = hv * Ws2[tid];
    #pragma unroll
    for (int off = 16; off > 0; off >>= 1) p += __shfl_xor(p, off, 32);
    if (lane == 0) red[wave] = p;
    __syncthreads();
    if (tid == 0) {
        float t = 0.f;
        for (int i = 0; i < 8; ++i) t += red[i];
        out[N] = t + bs2[0];
    }
}

// =====================================================================
extern "C" void kernel_launch(void* const* d_in, const int* in_sizes, int n_in,
                              void* d_out, int out_size, void* d_ws, size_t ws_size,
                              hipStream_t stream) {
    const float* node_features = (const float*)d_in[0];
    const float* adj    = (const float*)d_in[1];
    const float* claim  = (const float*)d_in[2];
    const int*   cur    = (const int*)d_in[3];
    const float* mask   = (const float*)d_in[4];
    const float* W1  = (const float*)d_in[5];
    const float* a1  = (const float*)d_in[6];
    const float* W2  = (const float*)d_in[7];
    const float* a2  = (const float*)d_in[8];
    const float* ln1g = (const float*)d_in[9];
    const float* ln1b = (const float*)d_in[10];
    const float* ln2g = (const float*)d_in[11];
    const float* ln2b = (const float*)d_in[12];
    const float* Wc  = (const float*)d_in[13];
    const float* bc  = (const float*)d_in[14];
    const float* Wa1 = (const float*)d_in[15];
    const float* ba1 = (const float*)d_in[16];
    const float* Wa2 = (const float*)d_in[17];
    const float* ba2 = (const float*)d_in[18];
    const float* Ws1 = (const float*)d_in[19];
    const float* bs1 = (const float*)d_in[20];
    const float* Ws2 = (const float*)d_in[21];
    const float* bs2 = (const float*)d_in[22];
    float* out = (float*)d_out;

    float* ws = (float*)d_ws;
    size_t o = 0;
    float* Wh   = ws + o; o += (size_t)HEADS * N * HID;   // 4M
    float* G    = ws + o; o += (size_t)N * D;             // 4M
    float* H1   = ws + o; o += (size_t)N * D;             // 4M
    float* H2   = ws + o; o += (size_t)N * D;             // 4M
    float* srcb = ws + o; o += (size_t)HEADS * N;
    float* dstb = ws + o; o += (size_t)HEADS * N;
    float* cp   = ws + o; o += D;
    float* vag  = ws + o; o += D;
    float* hid  = ws + o; o += (size_t)N * HID;           // 1M

    dim3 gGemm(N / 64, HID / 128, HEADS);

    // ---- GAT layer 1 ----
    k_gemm_wh<<<gGemm, 256, 0, stream>>>(node_features, NODE_DIM, W1, Wh);
    k_src_dst<<<dim3(N / 8, HEADS), 256, 0, stream>>>(Wh, a1, srcb, dstb);
    k_attn<<<dim3(N / 32, HEADS), 256, 0, stream>>>(Wh, srcb, dstb, adj, G);
    k_ln_elu<<<N, 256, 0, stream>>>(G, ln1g, ln1b, H1);

    // ---- GAT layer 2 ----
    k_gemm_wh<<<gGemm, 256, 0, stream>>>(H1, D, W2, Wh);
    k_src_dst<<<dim3(N / 8, HEADS), 256, 0, stream>>>(Wh, a2, srcb, dstb);
    k_attn<<<dim3(N / 32, HEADS), 256, 0, stream>>>(Wh, srcb, dstb, adj, G);
    k_ln_elu<<<N, 256, 0, stream>>>(G, ln2g, ln2b, H2);

    // ---- heads ----
    k_claim_visited<<<D / 256, 256, 0, stream>>>(claim, Wc, bc, H2, mask, cp, vag);
    k_edge_gemm<<<dim3(N / 64, HID / 128), 256, 0, stream>>>(cp, H2, vag, Wa1, ba1, hid);
    k_edge_final<<<N / 256, 256, 0, stream>>>(hid, Wa2, ba2, adj, cur, out);
    k_stop<<<1, 256, 0, stream>>>(cp, H2, vag, cur, Ws1, bs1, Ws2, bs2, out);
}
